// SelfAttentionLayer_35287451304614
// MI455X (gfx1250) — compile-verified, matrix operand order checked by reading, not run on hardware
//
#include <hip/hip_runtime.h>
#include <hip/hip_bf16.h>

typedef __attribute__((ext_vector_type(16))) _Float16 v16h;
typedef __attribute__((ext_vector_type(8)))  _Float16 v8h;
typedef __attribute__((ext_vector_type(4)))  _Float16 v4h;
typedef __attribute__((ext_vector_type(8)))  float    v8f;

#define B_  8
#define S_  2048
#define D_  1024
#define BS_ (B_*S_)

static __device__ __forceinline__ v16h mk_frag(const _Float16* p) {
  v8h lo = *(const v8h*)p;
  v8h hi = *(const v8h*)(p + 16);
  v16h f;
  #pragma unroll
  for (int i = 0; i < 8; ++i) { f[i] = lo[i]; f[i + 8] = hi[i]; }
  return f;
}

// Async global->LDS fill: nbytes128 16-byte packets, strided across nthreads.
// ldsByteOff is the packet's byte offset inside the dynamic LDS region, which
// starts at LDS address 0 (no static __shared__ in these kernels).
static __device__ __forceinline__ void async_stage(const void* gsrc, unsigned ldsBase,
                                                   int packets, int tid, int nthreads) {
  for (int i = tid; i < packets; i += nthreads) {
    unsigned loff = ldsBase + (unsigned)i * 16u;
    unsigned long long ga = (unsigned long long)(uintptr_t)((const char*)gsrc + (size_t)i * 16);
    asm volatile("global_load_async_to_lds_b128 %0, %1, off"
                 :: "v"(loff), "v"(ga) : "memory");
  }
  asm volatile("s_wait_asynccnt 0x0" ::: "memory");
}

// ---------------- kernel 1: convert x (f32) -> xb (f16) ----------------
__global__ void cvt_x_f16(const float* __restrict__ src, _Float16* __restrict__ dst, int n) {
  int i = (blockIdx.x * blockDim.x + threadIdx.x) * 4;
  if (i + 3 < n) {
    float4 f = *(const float4*)(src + i);
    v4h h = { (_Float16)f.x, (_Float16)f.y, (_Float16)f.z, (_Float16)f.w };
    *(v4h*)(dst + i) = h;
  }
}

// ------------- kernel 2a: transpose + convert Wq/Wk to f16 -------------
__global__ void prep_wT(const float* __restrict__ Wq, const float* __restrict__ Wk,
                        _Float16* __restrict__ WqT, _Float16* __restrict__ WkT) {
  int idx = blockIdx.x * blockDim.x + threadIdx.x;      // 0 .. D*D-1
  const float*  W  = blockIdx.y ? Wk  : Wq;
  _Float16*     WT = blockIdx.y ? WkT : WqT;
  int k = idx / D_, n = idx % D_;                       // read contiguous
  WT[(size_t)n * D_ + k] = (_Float16)W[(size_t)k * D_ + n];
}

// ----- kernel 2b: wvec = Wv @ Wo (f32 [D]),  c0 = bv . Wo (scalar) -----
__global__ void prep_wv(const float* __restrict__ Wv, const float* __restrict__ Wo,
                        const float* __restrict__ bv,
                        float* __restrict__ wvec, float* __restrict__ c0) {
  int d = blockIdx.x * blockDim.x + threadIdx.x;
  if (d < D_) {
    float s = 0.f;
    for (int j = 0; j < D_; ++j) s += Wv[(size_t)d * D_ + j] * Wo[j];
    wvec[d] = s;
  }
  if (blockIdx.x == 0 && threadIdx.x == 0) {
    float s = 0.f;
    for (int j = 0; j < D_; ++j) s += bv[j] * Wo[j];
    *c0 = s;
  }
}

// --------- kernel 3: Q/K projections. Block async-stages a 64-col slice of
// W^T in LDS (128 KB); each wave computes a 32x64 tile (2x4 WMMA blocking).
__global__ void qk_gemm(const _Float16* __restrict__ xb,
                        const _Float16* __restrict__ WqT, const _Float16* __restrict__ WkT,
                        const float* __restrict__ bq, const float* __restrict__ bk,
                        _Float16* __restrict__ qh, _Float16* __restrict__ kh) {
  extern __shared__ _Float16 sw[];                      // 64 * D halves = 128 KB
  const int lane = threadIdx.x & 31;
  const int wave = threadIdx.x >> 5;
  const int nColGrp = D_ / 64;                          // 16
  const int colGrp  = blockIdx.x % nColGrp;
  const int rowBlk  = blockIdx.x / nColGrp;             // 0..63
  const int colBase = colGrp * 64;
  const int rowBase = (rowBlk * 8 + wave) * 32;

  const _Float16* WT   = blockIdx.y ? WkT : WqT;
  const float*    bias = blockIdx.y ? bk  : bq;
  _Float16*       out  = blockIdx.y ? kh  : qh;

  // async global->LDS stage of W^T[colBase .. colBase+63][:] (contiguous 128 KB)
  async_stage(WT + (size_t)colBase * D_, 0u, 64 * D_ * 2 / 16, threadIdx.x, 256);
  __syncthreads();

  const int half = lane >> 4;
  const int mn   = lane & 15;
  const _Float16* arow0 = xb + (size_t)(rowBase + mn) * D_ + half * 8;
  const _Float16* arow1 = arow0 + (size_t)16 * D_;
  const _Float16* brow[4];
  #pragma unroll
  for (int j = 0; j < 4; ++j) brow[j] = sw + (size_t)(j * 16 + mn) * D_ + half * 8;

  v8f acc[2][4] = {};
  #pragma unroll 2
  for (int k0 = 0; k0 < D_; k0 += 32) {
    v16h a0 = mk_frag(arow0 + k0);
    v16h a1 = mk_frag(arow1 + k0);
    v16h b0 = mk_frag(brow[0] + k0);
    v16h b1 = mk_frag(brow[1] + k0);
    v16h b2 = mk_frag(brow[2] + k0);
    v16h b3 = mk_frag(brow[3] + k0);
    acc[0][0] = __builtin_amdgcn_wmma_f32_16x16x32_f16(false, a0, false, b0, (short)0, acc[0][0], false, false);
    acc[0][1] = __builtin_amdgcn_wmma_f32_16x16x32_f16(false, a0, false, b1, (short)0, acc[0][1], false, false);
    acc[0][2] = __builtin_amdgcn_wmma_f32_16x16x32_f16(false, a0, false, b2, (short)0, acc[0][2], false, false);
    acc[0][3] = __builtin_amdgcn_wmma_f32_16x16x32_f16(false, a0, false, b3, (short)0, acc[0][3], false, false);
    acc[1][0] = __builtin_amdgcn_wmma_f32_16x16x32_f16(false, a1, false, b0, (short)0, acc[1][0], false, false);
    acc[1][1] = __builtin_amdgcn_wmma_f32_16x16x32_f16(false, a1, false, b1, (short)0, acc[1][1], false, false);
    acc[1][2] = __builtin_amdgcn_wmma_f32_16x16x32_f16(false, a1, false, b2, (short)0, acc[1][2], false, false);
    acc[1][3] = __builtin_amdgcn_wmma_f32_16x16x32_f16(false, a1, false, b3, (short)0, acc[1][3], false, false);
  }
  #pragma unroll
  for (int j = 0; j < 4; ++j) {
    float bb = bias[colBase + j * 16 + mn];
    #pragma unroll
    for (int i = 0; i < 2; ++i) {
      #pragma unroll
      for (int r = 0; r < 8; ++r) {
        int m = i * 16 + r + half * 8;
        out[(size_t)(rowBase + m) * D_ + colBase + j * 16 + mn] = (_Float16)(acc[i][j][r] + bb);
      }
    }
  }
}

// ---------------- kernel 4: vw[row] = x[row,:].wvec + c0 ----------------
__global__ void vw_dot(const float* __restrict__ x, const float* __restrict__ wvec,
                       const float* __restrict__ c0, float* __restrict__ vw) {
  const int lane = threadIdx.x & 31;
  const int wave = threadIdx.x >> 5;
  const int row  = blockIdx.x * 8 + wave;
  const float* xr = x + (size_t)row * D_;
  float s = 0.f;
  #pragma unroll
  for (int d = lane * 4; d < D_; d += 128) {
    float4 f = *(const float4*)(xr + d);
    s += f.x * wvec[d] + f.y * wvec[d+1] + f.z * wvec[d+2] + f.w * wvec[d+3];
  }
  #pragma unroll
  for (int off = 16; off; off >>= 1) s += __shfl_xor(s, off, 32);
  if (lane == 0) vw[row] = s + *c0;
}

// ------ kernel 5: scores via WMMA + flash-style online softmax + vw sum ------
// Block = 32 queries (q slice async-staged once in LDS, shared by 4 waves).
// Wave w owns key chunk [w*512, w*512+512); chunks merged via LDS log-sum-exp.
// D-tile = K_tile(A) x Q_tile(B): query = lane&15, keys = acc VGPR idx (+8 hi half).
__global__ void attn(const _Float16* __restrict__ qh, const _Float16* __restrict__ kh,
                     const float* __restrict__ vw, const float* __restrict__ bo,
                     float* __restrict__ outp) {
  extern __shared__ _Float16 smem[];                    // 32*D halves + reduce area
  const int lane = threadIdx.x & 31;
  const int wave = threadIdx.x >> 5;                    // = key chunk 0..3
  const int b    = blockIdx.x / (S_ / 32);
  const int qg   = blockIdx.x % (S_ / 32);
  const size_t qrow0 = (size_t)b * S_ + qg * 32;
  float* red = (float*)(smem + 32 * D_);                // [4][32][3]

  // async global->LDS stage of 32 query rows (64 KB)
  async_stage(qh + qrow0 * D_, 0u, 32 * D_ * 2 / 16, threadIdx.x, 128);
  __syncthreads();

  const int half = lane >> 4;
  const int mn   = lane & 15;
  const _Float16* qb0 = smem + (size_t)mn * D_ + half * 8;        // q tile 0
  const _Float16* qb1 = smem + (size_t)(16 + mn) * D_ + half * 8; // q tile 1

  float m0 = -__builtin_inff(), l0 = 0.f, a0s = 0.f;    // state for q tile 0
  float m1 = -__builtin_inff(), l1 = 0.f, a1s = 0.f;    // state for q tile 1

  const int kStart = wave * (S_ / 4);                   // 512-key chunk
  for (int kt = 0; kt < S_ / 4; kt += 32) {
    const size_t krow0 = (size_t)b * S_ + kStart + kt;
    const _Float16* ka0 = kh + (krow0 + mn) * D_ + half * 8;
    const _Float16* ka1 = ka0 + (size_t)16 * D_;
    v8f c00 = {}, c01 = {}, c10 = {}, c11 = {};
    #pragma unroll 2
    for (int k0 = 0; k0 < D_; k0 += 32) {
      v16h A0 = mk_frag(ka0 + k0);
      v16h A1 = mk_frag(ka1 + k0);
      v16h Bq0 = mk_frag(qb0 + k0);
      v16h Bq1 = mk_frag(qb1 + k0);
      c00 = __builtin_amdgcn_wmma_f32_16x16x32_f16(false, A0, false, Bq0, (short)0, c00, false, false);
      c01 = __builtin_amdgcn_wmma_f32_16x16x32_f16(false, A0, false, Bq1, (short)0, c01, false, false);
      c10 = __builtin_amdgcn_wmma_f32_16x16x32_f16(false, A1, false, Bq0, (short)0, c10, false, false);
      c11 = __builtin_amdgcn_wmma_f32_16x16x32_f16(false, A1, false, Bq1, (short)0, c11, false, false);
    }
    // vw for the 32 keys of this step
    const float* vp = vw + krow0 + half * 8;
    float4 v0 = *(const float4*)vp;
    float4 v1 = *(const float4*)(vp + 4);
    float4 v2 = *(const float4*)(vp + 16);
    float4 v3 = *(const float4*)(vp + 20);
    float vA[8] = { v0.x, v0.y, v0.z, v0.w, v1.x, v1.y, v1.z, v1.w };
    float vB[8] = { v2.x, v2.y, v2.z, v2.w, v3.x, v3.y, v3.z, v3.w };

    // online softmax update for q tile 0 (16 key scores: c00 + c10)
    {
      float t = fmaxf(c00[0], c10[0]);
      #pragma unroll
      for (int r = 1; r < 8; ++r) t = fmaxf(t, fmaxf(c00[r], c10[r]));
      float nm = fmaxf(m0, t);
      float sc = __expf(m0 - nm);
      l0 *= sc; a0s *= sc;
      #pragma unroll
      for (int r = 0; r < 8; ++r) {
        float e0 = __expf(c00[r] - nm);
        float e1 = __expf(c10[r] - nm);
        l0 += e0 + e1; a0s += e0 * vA[r] + e1 * vB[r];
      }
      m0 = nm;
    }
    // online softmax update for q tile 1 (c01 + c11)
    {
      float t = fmaxf(c01[0], c11[0]);
      #pragma unroll
      for (int r = 1; r < 8; ++r) t = fmaxf(t, fmaxf(c01[r], c11[r]));
      float nm = fmaxf(m1, t);
      float sc = __expf(m1 - nm);
      l1 *= sc; a1s *= sc;
      #pragma unroll
      for (int r = 0; r < 8; ++r) {
        float e0 = __expf(c01[r] - nm);
        float e1 = __expf(c11[r] - nm);
        l1 += e0 + e1; a1s += e0 * vA[r] + e1 * vB[r];
      }
      m1 = nm;
    }
  }

  // merge lane halves (lane n holds keys r, lane n+16 holds keys r+8)
  {
    float mm = __shfl_xor(m0, 16, 32), ll = __shfl_xor(l0, 16, 32), aa = __shfl_xor(a0s, 16, 32);
    float M = fmaxf(m0, mm);
    float L = l0 * __expf(m0 - M) + ll * __expf(mm - M);
    float A = a0s * __expf(m0 - M) + aa * __expf(mm - M);
    if (half == 0) { float* r = red + (wave * 32 + mn) * 3; r[0] = M; r[1] = L; r[2] = A; }
  }
  {
    float mm = __shfl_xor(m1, 16, 32), ll = __shfl_xor(l1, 16, 32), aa = __shfl_xor(a1s, 16, 32);
    float M = fmaxf(m1, mm);
    float L = l1 * __expf(m1 - M) + ll * __expf(mm - M);
    float A = a1s * __expf(m1 - M) + aa * __expf(mm - M);
    if (half == 0) { float* r = red + (wave * 32 + 16 + mn) * 3; r[0] = M; r[1] = L; r[2] = A; }
  }
  __syncthreads();

  // combine 4 key chunks per query; threads 0..31 write the 32 outputs
  if (threadIdx.x < 32) {
    int q = threadIdx.x;
    float M = -__builtin_inff();
    #pragma unroll
    for (int ck = 0; ck < 4; ++ck) M = fmaxf(M, red[(ck * 32 + q) * 3 + 0]);
    float L = 0.f, A = 0.f;
    #pragma unroll
    for (int ck = 0; ck < 4; ++ck) {
      const float* r = red + (ck * 32 + q) * 3;
      float e = __expf(r[0] - M);
      L += r[1] * e; A += r[2] * e;
    }
    outp[qrow0 + q] = A / L + bo[0];
  }
}

// ------------------------------ launcher ------------------------------
extern "C" void kernel_launch(void* const* d_in, const int* in_sizes, int n_in,
                              void* d_out, int out_size, void* d_ws, size_t ws_size,
                              hipStream_t stream) {
  (void)in_sizes; (void)n_in; (void)out_size; (void)ws_size;
  const float* x  = (const float*)d_in[0];
  const float* Wq = (const float*)d_in[1];
  const float* bq = (const float*)d_in[2];
  const float* Wk = (const float*)d_in[3];
  const float* bk = (const float*)d_in[4];
  const float* Wv = (const float*)d_in[5];
  const float* bv = (const float*)d_in[6];
  const float* Wo = (const float*)d_in[7];
  const float* bo = (const float*)d_in[8];
  float* out = (float*)d_out;

  char* ws = (char*)d_ws;
  size_t off = 0;
  auto alloc = [&](size_t bytes) { char* p = ws + off; off = (off + bytes + 255) & ~(size_t)255; return p; };
  _Float16* xb   = (_Float16*)alloc((size_t)BS_ * D_ * 2);
  _Float16* WqT  = (_Float16*)alloc((size_t)D_ * D_ * 2);
  _Float16* WkT  = (_Float16*)alloc((size_t)D_ * D_ * 2);
  _Float16* qh   = (_Float16*)alloc((size_t)BS_ * D_ * 2);
  _Float16* kh   = (_Float16*)alloc((size_t)BS_ * D_ * 2);
  float*    wvec = (float*)alloc((size_t)D_ * 4);
  float*    c0   = (float*)alloc(256);
  float*    vwb  = (float*)alloc((size_t)BS_ * 4);

  const int n_x = BS_ * D_;
  cvt_x_f16<<<(n_x / 4 + 255) / 256, 256, 0, stream>>>(x, xb, n_x);
  prep_wT<<<dim3(D_ * D_ / 256, 2), 256, 0, stream>>>(Wq, Wk, WqT, WkT);
  prep_wv<<<(D_ + 255) / 256, 256, 0, stream>>>(Wv, Wo, bv, wvec, c0);
  // qk_gemm: 64 row-blocks (8 waves x 32 rows) x 16 col-groups, y: 0=q 1=k
  qk_gemm<<<dim3(64 * (D_ / 64), 2), 256, (size_t)64 * D_ * 2, stream>>>(
      xb, WqT, WkT, bq, bk, qh, kh);
  vw_dot<<<BS_ / 8, 256, 0, stream>>>(x, wvec, c0, vwb);
  // attn: one block per 32 queries; 4 waves = 4 key chunks
  attn<<<BS_ / 32, 128, (size_t)32 * D_ * 2 + 4 * 32 * 3 * 4, stream>>>(
      qh, kh, vwb, bo, out);
}
